// PPN_75660143886362
// MI455X (gfx1250) — matrix operand order repaired and checked by reading, not compiled, not run on hardware
//
#include <hip/hip_runtime.h>
#include <hip/hip_bf16.h>
#include <math.h>

// ---------- WMMA fragment types (wave32, gfx1250) ----------
typedef __attribute__((ext_vector_type(16))) _Float16 v16h;
typedef __attribute__((ext_vector_type(8)))  _Float16 v8h;
typedef __attribute__((ext_vector_type(8)))  float    v8f;

#define LOGIT4 1.3862943611198906f   // sigmoid(x) > 0.8  <=>  x > ln(4)
#define N128   2097152               // 128^3

__device__ __forceinline__ float eluf(float x) {
  return x > 0.f ? x : (expf(x) - 1.f);
}

// ============================================================
// K1: x = conv3(fm1, w1c)   80ch, 8^3, SAME   (tiny, scalar)
// ============================================================
__global__ void k1_conv80(const float* __restrict__ fm1,
                          const float* __restrict__ w1c,
                          float* __restrict__ x80) {
  int tid = blockIdx.x * blockDim.x + threadIdx.x;
  if (tid >= 80 * 512) return;
  int c = tid >> 9, v = tid & 511;
  int d = v >> 6, h = (v >> 3) & 7, w = v & 7;
  float acc = 0.f;
  for (int t = 0; t < 27; ++t) {
    int zz = d + t / 9 - 1, yy = h + (t / 3) % 3 - 1, xx = w + t % 3 - 1;
    if ((unsigned)zz >= 8u || (unsigned)yy >= 8u || (unsigned)xx >= 8u) continue;
    int nv = (zz << 6) + (yy << 3) + xx;
    const float* wp = w1c + c * 2160 + t;   // [c][ci][tap]
    const float* ip = fm1 + nv;
    for (int ci = 0; ci < 80; ++ci)
      acc += ip[ci << 9] * wp[ci * 27];
  }
  x80[tid] = acc;
}

// ============================================================
// K2: scores = conv3(x, w1s); mask8 = sigmoid>THR;
//     h1a = elu(x*bn1a_s + bn1a_b) * mask
// ============================================================
__global__ void k2_score_mask(const float* __restrict__ x80,
                              const float* __restrict__ w1s,
                              const float* __restrict__ s,
                              const float* __restrict__ b,
                              float* __restrict__ h1a,
                              unsigned char* __restrict__ mask8) {
  int v = blockIdx.x * blockDim.x + threadIdx.x;
  if (v >= 512) return;
  int d = v >> 6, h = (v >> 3) & 7, w = v & 7;
  float sc = 0.f;
  for (int t = 0; t < 27; ++t) {
    int zz = d + t / 9 - 1, yy = h + (t / 3) % 3 - 1, xx = w + t % 3 - 1;
    if ((unsigned)zz >= 8u || (unsigned)yy >= 8u || (unsigned)xx >= 8u) continue;
    int nv = (zz << 6) + (yy << 3) + xx;
    for (int ci = 0; ci < 80; ++ci)
      sc += x80[(ci << 9) + nv] * w1s[ci * 27 + t];
  }
  bool m = sc > LOGIT4;
  mask8[v] = m ? 1 : 0;
  for (int c = 0; c < 80; ++c)
    h1a[(c << 9) + v] = m ? eluf(x80[(c << 9) + v] * s[c] + b[c]) : 0.f;
}

// ============================================================
// K3a: tconv 80->16 (k2 s2) to 16^3, then BN+ELU, masked
// ============================================================
__global__ void k3a_tconv1(const float* __restrict__ h1a,
                           const unsigned char* __restrict__ mask8,
                           const float* __restrict__ wt1a,  // [80][16][2][2][2]
                           const float* __restrict__ s,
                           const float* __restrict__ b,
                           float* __restrict__ h1b) {
  int tid = blockIdx.x * blockDim.x + threadIdx.x;
  if (tid >= 16 * 4096) return;
  int co = tid >> 12, v = tid & 4095;
  int d = v >> 8, h = (v >> 4) & 15, w = v & 15;
  int p = ((d >> 1) << 6) + ((h >> 1) << 3) + (w >> 1);
  int par = ((d & 1) << 2) | ((h & 1) << 1) | (w & 1);
  float r = 0.f;
  if (mask8[p]) {
    float t = 0.f;
    for (int ci = 0; ci < 80; ++ci)
      t += h1a[(ci << 9) + p] * wt1a[ci * 128 + co * 8 + par];
    r = eluf(t * s[co] + b[co]);
  }
  h1b[(co << 12) + v] = r;
}

// ============================================================
// K3b: tconv 16->1 to 32^3, attn1, keep1; h2a = elu(fm2*bn+..)*keep1
// ============================================================
__global__ void k3b_keep1(const float* __restrict__ h1b,
                          const unsigned char* __restrict__ mask8,
                          const float* __restrict__ wt1b,  // [16][1][2][2][2]
                          const float* __restrict__ fm2,
                          const float* __restrict__ s,
                          const float* __restrict__ b,
                          float* __restrict__ h2a,
                          unsigned char* __restrict__ keep1) {
  int v = blockIdx.x * blockDim.x + threadIdx.x;
  if (v >= 32768) return;
  int d = v >> 10, h = (v >> 5) & 31, w = v & 31;
  int p16 = ((d >> 1) << 8) + ((h >> 1) << 4) + (w >> 1);
  int par = ((d & 1) << 2) | ((h & 1) << 1) | (w & 1);
  bool m32 = mask8[((d >> 2) << 6) + ((h >> 2) << 3) + (w >> 2)];
  float t = 0.f;
  for (int ci = 0; ci < 16; ++ci)
    t += h1b[(ci << 12) + p16] * wt1b[ci * 8 + par];
  bool kp = m32 && (t > LOGIT4);
  if (kp) {
    bool any = false;
    for (int c = 0; c < 48; ++c) any = any || (fm2[c * 32768 + v] > 0.f);
    kp = any;
  }
  keep1[v] = kp ? 1 : 0;
  for (int c = 0; c < 48; ++c)
    h2a[c * 32768 + v] = kp ? eluf(fm2[c * 32768 + v] * s[c] + b[c]) : 0.f;
}

// ============================================================
// K4: tconv 48->64 to 64^3, BN+ELU, masked
// ============================================================
__global__ void k4_tconv2(const float* __restrict__ h2a,
                          const unsigned char* __restrict__ keep1,
                          const float* __restrict__ wt2a,  // [48][64][2][2][2]
                          const float* __restrict__ s,
                          const float* __restrict__ b,
                          float* __restrict__ h2b) {
  int v = blockIdx.x * blockDim.x + threadIdx.x;
  if (v >= 262144) return;
  int d = v >> 12, h = (v >> 6) & 63, w = v & 63;
  int p = ((d >> 1) << 10) + ((h >> 1) << 5) + (w >> 1);
  int par = ((d & 1) << 2) | ((h & 1) << 1) | (w & 1);
  if (!keep1[p]) {
    for (int co = 0; co < 64; ++co) h2b[co * 262144 + v] = 0.f;
    return;
  }
  float a48[48];
  for (int ci = 0; ci < 48; ++ci) a48[ci] = h2a[ci * 32768 + p];
  for (int co = 0; co < 64; ++co) {
    float t = 0.f;
    for (int ci = 0; ci < 48; ++ci)
      t += a48[ci] * wt2a[ci * 512 + co * 8 + par];
    h2b[co * 262144 + v] = eluf(t * s[co] + b[co]);
  }
}

// ============================================================
// K5: tconv 64->1 to 128^3, attn2, keep2; build masked f16
//     NDHWC z buffer; emit output channels 0..3 (bidx + grid*k2)
// ============================================================
__global__ void k5_keep2(const float* __restrict__ h2b,
                         const unsigned char* __restrict__ keep1,
                         const float* __restrict__ wt2b,  // [64][1][2][2][2]
                         const float* __restrict__ fm3,
                         unsigned char* __restrict__ keep2,
                         _Float16* __restrict__ zin,
                         float* __restrict__ out) {
  int v = blockIdx.x * blockDim.x + threadIdx.x;
  if (v >= N128) return;
  int d = v >> 14, h = (v >> 7) & 127, w = v & 127;
  int p = ((d >> 1) << 12) + ((h >> 1) << 6) + (w >> 1);
  int par = ((d & 1) << 2) | ((h & 1) << 1) | (w & 1);
  bool k1u = keep1[((d >> 2) << 10) + ((h >> 2) << 5) + (w >> 2)];
  float t = 0.f;
  for (int ci = 0; ci < 64; ++ci)
    t += h2b[ci * 262144 + p] * wt2b[ci * 8 + par];
  bool kp = k1u && (t > LOGIT4);
  if (kp) {
    bool any = false;
    for (int c = 0; c < 16; ++c) any = any || (fm3[c * N128 + v] > 0.f);
    kp = any;
  }
  keep2[v] = kp ? 1 : 0;
  for (int c = 0; c < 16; ++c)
    zin[v * 16 + c] = kp ? (_Float16)fm3[c * N128 + v] : (_Float16)0.f;
  float kf = kp ? 1.f : 0.f;
  out[v]            = 0.f;             // batch index column
  out[N128 + v]     = (float)d * kf;   // grid[0]*k2
  out[2 * N128 + v] = (float)h * kf;   // grid[1]*k2
  out[3 * N128 + v] = (float)w * kf;   // grid[2]*k2
}

// ============================================================
// WMMA implicit-GEMM 3x3x3 conv, 16 in-ch, 16 out-ch, 128^3.
// One wave computes a 16-voxel x-row segment x 16 out-channels.
// K = 27 taps * 16 ch, packed 2 taps per v_wmma_f32_16x16x32_f16.
//
// Fragment layouts (ISA 7.12.2, wave32):
//  A: lane l -> row M=l&15; elems 0..7 = K 8*(l>>4)+0..7   (tap A)
//               elems 8..15 = K 16+8*(l>>4)+0..7           (tap B)
//  B: lane l -> col N=l&15; elems 0..7 = K 8*(l>>4)+0..7   (tap A)
//     elems 8..15 = K 16+8*(l>>4)+0..7 (tap B); in-ch = K mod 16
//  D: lane l -> col N=l&15, rows M = e + 8*(l>>4)
// ============================================================
__device__ __forceinline__ v8f conv_tile_wmma(const _Float16* __restrict__ zsrc,
                                              const _Float16* __restrict__ wB,
                                              int zi, int yi, int xbase,
                                              int lane) {
  int mrow = lane & 15;
  int hi = lane >> 4;
  v8f acc = {};
  for (int tp = 0; tp < 14; ++tp) {
    int tA = 2 * tp, tB = 2 * tp + 1;
    v8h alo = {}, ahi = {};
    {
      int zz = zi + tA / 9 - 1, yy = yi + (tA / 3) % 3 - 1;
      int xx = xbase + mrow + tA % 3 - 1;
      if ((unsigned)zz < 128u && (unsigned)yy < 128u && (unsigned)xx < 128u) {
        int vox = (zz << 14) + (yy << 7) + xx;
        alo = *(const v8h*)(zsrc + vox * 16 + hi * 8);   // global_load_b128
      }
    }
    if (tB < 27) {
      int zz = zi + tB / 9 - 1, yy = yi + (tB / 3) % 3 - 1;
      int xx = xbase + mrow + tB % 3 - 1;
      if ((unsigned)zz < 128u && (unsigned)yy < 128u && (unsigned)xx < 128u) {
        int vox = (zz << 14) + (yy << 7) + xx;
        ahi = *(const v8h*)(zsrc + vox * 16 + hi * 8);
      }
    }
    v16h a;
#pragma unroll
    for (int q = 0; q < 8; ++q) { a[q] = alo[q]; a[q + 8] = ahi[q]; }

    // elems 0..7: tap A ch 8*hi+q, elems 8..15: tap B ch 8*hi+q
    v16h bf = {};
    {
      v8h b0 = *(const v8h*)(wB + tA * 256 + (lane & 15) * 16 + hi * 8);
      v8h b1 = {};
      if (tB < 27) b1 = *(const v8h*)(wB + tB * 256 + (lane & 15) * 16 + hi * 8);
#pragma unroll
      for (int q = 0; q < 8; ++q) { bf[q] = b0[q]; bf[q + 8] = b1[q]; }
    }
    acc = __builtin_amdgcn_wmma_f32_16x16x32_f16(
        false, a, false, bf, (short)0, acc, false, false);
  }
  return acc;
}

// K6: zc = conv3(z, w3c) * k2     (f16 NDHWC in, f16 NDHWC out)
__global__ void __launch_bounds__(256)
k6_conv_wmma(const _Float16* __restrict__ zin,
             const float* __restrict__ w3c,   // [16][16][27]
             const unsigned char* __restrict__ keep2,
             _Float16* __restrict__ zc) {
  __shared__ _Float16 wB[27 * 16 * 16];       // [tap][o][i]
  for (int idx = threadIdx.x; idx < 27 * 256; idx += 256) {
    int tap = idx >> 8, rem = idx & 255, o = rem >> 4, i = rem & 15;
    wB[idx] = (_Float16)w3c[o * 432 + i * 27 + tap];
  }
  __syncthreads();

  int zi = blockIdx.x >> 7, yi = blockIdx.x & 127;
  int lane = threadIdx.x & 31;
  int xbase = (threadIdx.x >> 5) << 4;

  v8f acc = conv_tile_wmma(zin, wB, zi, yi, xbase, lane);

  int n = lane & 15, hi = lane >> 4;
#pragma unroll
  for (int e = 0; e < 8; ++e) {
    int xo = xbase + e + 8 * hi;
    int vox = (zi << 14) + (yi << 7) + xo;
    float kf = keep2[vox] ? 1.f : 0.f;
    zc[vox * 16 + n] = (_Float16)(acc[e] * kf);
  }
}

// K7: heads pix(3)/sc(1)/ty(5) = conv3(zc, w3{p,s,t}) padded to 16
//     out-channels; adds anchors; writes output channels 4..12.
__global__ void __launch_bounds__(256)
k7_heads_wmma(const _Float16* __restrict__ zc,
              const float* __restrict__ w3p,  // [3][16][27]
              const float* __restrict__ w3s,  // [1][16][27]
              const float* __restrict__ w3t,  // [5][16][27]
              const unsigned char* __restrict__ keep2,
              float* __restrict__ out) {
  __shared__ _Float16 wB[27 * 16 * 16];       // [tap][o][i], o>=9 zero
  for (int idx = threadIdx.x; idx < 27 * 256; idx += 256) {
    int tap = idx >> 8, rem = idx & 255, o = rem >> 4, i = rem & 15;
    float wv = 0.f;
    if (o < 3)       wv = w3p[o * 432 + i * 27 + tap];
    else if (o == 3) wv = w3s[i * 27 + tap];
    else if (o < 9)  wv = w3t[(o - 4) * 432 + i * 27 + tap];
    wB[idx] = (_Float16)wv;
  }
  __syncthreads();

  int zi = blockIdx.x >> 7, yi = blockIdx.x & 127;
  int lane = threadIdx.x & 31;
  int xbase = (threadIdx.x >> 5) << 4;

  v8f acc = conv_tile_wmma(zc, wB, zi, yi, xbase, lane);

  int n = lane & 15, hi = lane >> 4;
  if (n < 9) {
#pragma unroll
    for (int e = 0; e < 8; ++e) {
      int xo = xbase + e + 8 * hi;
      int vox = (zi << 14) + (yi << 7) + xo;
      float kf = keep2[vox] ? 1.f : 0.f;
      float val = acc[e];
      int och;
      if (n < 3) {                // pix: add anchor (coord + 0.5) before mask
        float anchor = (n == 0 ? (float)zi : n == 1 ? (float)yi : (float)xo) + 0.5f;
        val += anchor;
        och = 4 + n;
      } else if (n == 3) {
        och = 7;                  // score head
      } else {
        och = 4 + n;              // ty heads: n=4..8 -> 8..12
      }
      out[och * N128 + vox] = val * kf;
    }
  }
}

// ============================================================
extern "C" void kernel_launch(void* const* d_in, const int* in_sizes, int n_in,
                              void* d_out, int out_size, void* d_ws, size_t ws_size,
                              hipStream_t stream) {
  const float* fm1    = (const float*)d_in[0];
  const float* fm2    = (const float*)d_in[1];
  const float* fm3    = (const float*)d_in[2];
  const float* w1c    = (const float*)d_in[3];
  const float* w1s    = (const float*)d_in[4];
  const float* bn1a_s = (const float*)d_in[5];
  const float* bn1a_b = (const float*)d_in[6];
  const float* wt1a   = (const float*)d_in[7];
  const float* bn1b_s = (const float*)d_in[8];
  const float* bn1b_b = (const float*)d_in[9];
  const float* wt1b   = (const float*)d_in[10];
  const float* bn2a_s = (const float*)d_in[11];
  const float* bn2a_b = (const float*)d_in[12];
  const float* wt2a   = (const float*)d_in[13];
  const float* bn2b_s = (const float*)d_in[14];
  const float* bn2b_b = (const float*)d_in[15];
  const float* wt2b   = (const float*)d_in[16];
  const float* w3c    = (const float*)d_in[17];
  const float* w3p    = (const float*)d_in[18];
  const float* w3s    = (const float*)d_in[19];
  const float* w3t    = (const float*)d_in[20];
  float* out = (float*)d_out;

  char* ws = (char*)d_ws;
  size_t off = 0;
  auto wsalloc = [&](size_t bytes) -> void* {
    void* p = (void*)(ws + off);
    off = (off + bytes + 255) & ~(size_t)255;
    return p;
  };
  float*         x80   = (float*)        wsalloc(80 * 512 * 4);
  float*         h1a   = (float*)        wsalloc(80 * 512 * 4);
  unsigned char* mask8 = (unsigned char*)wsalloc(512);
  float*         h1b   = (float*)        wsalloc(16 * 4096 * 4);
  unsigned char* keep1 = (unsigned char*)wsalloc(32768);
  float*         h2a   = (float*)        wsalloc((size_t)48 * 32768 * 4);
  float*         h2b   = (float*)        wsalloc((size_t)64 * 262144 * 4);
  unsigned char* keep2 = (unsigned char*)wsalloc(N128);
  _Float16*      zin   = (_Float16*)     wsalloc((size_t)N128 * 16 * 2);
  _Float16*      zc    = (_Float16*)     wsalloc((size_t)N128 * 16 * 2);

  k1_conv80   <<<160,  256, 0, stream>>>(fm1, w1c, x80);
  k2_score_mask<<<2,   256, 0, stream>>>(x80, w1s, bn1a_s, bn1a_b, h1a, mask8);
  k3a_tconv1  <<<256,  256, 0, stream>>>(h1a, mask8, wt1a, bn1b_s, bn1b_b, h1b);
  k3b_keep1   <<<128,  256, 0, stream>>>(h1b, mask8, wt1b, fm2, bn2a_s, bn2a_b, h2a, keep1);
  k4_tconv2   <<<1024, 256, 0, stream>>>(h2a, keep1, wt2a, bn2b_s, bn2b_b, h2b);
  k5_keep2    <<<8192, 256, 0, stream>>>(h2b, keep1, wt2b, fm3, keep2, zin, out);
  k6_conv_wmma<<<16384,256, 0, stream>>>(zin, w3c, keep2, zc);
  k7_heads_wmma<<<16384,256, 0, stream>>>(zc, w3p, w3s, w3t, keep2, out);
}
